// MACE_layer_8452495639067
// MI455X (gfx1250) — matrix operand order repaired and checked by reading, not compiled, not run on hardware
//
#include <hip/hip_runtime.h>
#include <hip/hip_bf16.h>

typedef __attribute__((ext_vector_type(16))) __bf16 v16bf;
typedef __attribute__((ext_vector_type(8)))  float  v8f;

#define NNODES 20000
#define NEDGES 320000

__device__ __forceinline__ __bf16 f2bf(float f) {
    union { float f; unsigned u; } in; in.f = f;
    unsigned u = in.u;
    unsigned r = u + 0x7FFFu + ((u >> 16) & 1u);       // round-to-nearest-even
    union { unsigned short s; __bf16 b; } out;
    out.s = (unsigned short)(r >> 16);
    return out.b;
}

__device__ __forceinline__ float silu(float x) {
    return x / (1.0f + __expf(-x));
}

// ---------------------------------------------------------------------------
// Weight pre-swizzle into WMMA B-operand lane layout (bf16).
// B 32x16 tile: lane l -> col n = nt*16 + (l&15); half = l>>4; elem e -> k = (e&7) + 16*(e>>3) + 8*half
// ---------------------------------------------------------------------------
__global__ void prep_weights(const float* __restrict__ W_up, const float* __restrict__ Wd0,
                             const float* __restrict__ Wd1, const float* __restrict__ Wp0,
                             const float* __restrict__ Wp1, const float* __restrict__ rw1,
                             const float* __restrict__ rw2, __bf16* __restrict__ wb) {
    int tid = blockIdx.x * blockDim.x + threadIdx.x;
    int nth = gridDim.x * blockDim.x;
    const float* srcs[5] = {W_up, Wd0, Wd1, Wp0, Wp1};
    // five 128x128 matrices: kt(4) x nt(8) x lane(32) x e(16)
    for (int wi = 0; wi < 5; ++wi) {
        __bf16* dst = wb + wi * 16384;
        const float* src = srcs[wi];
        for (int idx = tid; idx < 16384; idx += nth) {
            int kt = idx >> 12, nt = (idx >> 9) & 7, l = (idx >> 4) & 31, e = idx & 15;
            int k = kt * 32 + (e & 7) + 16 * (e >> 3) + 8 * (l >> 4);
            int n = nt * 16 + (l & 15);
            dst[idx] = f2bf(src[k * 128 + n]);
        }
    }
    // radial_w1 (9x64) zero-padded to 32x64: nt(4) x lane(32) x e(16)
    {
        __bf16* dst = wb + 81920;
        for (int idx = tid; idx < 2048; idx += nth) {
            int nt = idx >> 9, l = (idx >> 4) & 31, e = idx & 15;
            int k = (e & 7) + 16 * (e >> 3) + 8 * (l >> 4);
            int n = nt * 16 + (l & 15);
            dst[idx] = f2bf(k < 9 ? rw1[k * 64 + n] : 0.0f);
        }
    }
    // radial_w2 (64x256): kt(2) x nt(16) x lane(32) x e(16)
    {
        __bf16* dst = wb + 83968;
        for (int idx = tid; idx < 16384; idx += nth) {
            int kt = idx >> 13, nt = (idx >> 9) & 15, l = (idx >> 4) & 31, e = idx & 15;
            int k = kt * 32 + (e & 7) + 16 * (e >> 3) + 8 * (l >> 4);
            int n = nt * 16 + (l & 15);
            dst[idx] = f2bf(rw2[k * 256 + n]);
        }
    }
}

__global__ void zero_kernel(float* __restrict__ p, int count) {
    int i = blockIdx.x * blockDim.x + threadIdx.x;
    int stride = gridDim.x * blockDim.x;
    for (; i < count; i += stride) p[i] = 0.0f;
}

// ---------------------------------------------------------------------------
// Generic Y(nrows x 128) = X(nrows x 128) @ W(128 x 128), bf16 WMMA, f32 accum.
// One wave per 16-row strip; 4 K-steps x 8 N-tiles = 32 v_wmma per wave.
// A 16x32 bf16 layout: lane l -> row m = l&15, half = l>>4; elem e -> k = (e&7)+16*(e>>3)+8*half
// ---------------------------------------------------------------------------
__global__ void gemm128(const float* __restrict__ X, const __bf16* __restrict__ Wsw,
                        float* __restrict__ Y, int nrows) {
    const int lane = threadIdx.x & 31;
    const int waveId = blockIdx.x * (blockDim.x >> 5) + (threadIdx.x >> 5);
    const int row0 = waveId * 16;
    if (row0 >= nrows) return;
    const int m = lane & 15;
    const int half = lane >> 4;
    const float* xr = X + (size_t)(row0 + m) * 128;

    v8f acc[8] = {};
#pragma unroll
    for (int kt = 0; kt < 4; ++kt) {
        const float* xp = xr + kt * 32;
        float4 f0 = *(const float4*)(xp + 8 * half);
        float4 f1 = *(const float4*)(xp + 8 * half + 4);
        float4 f2 = *(const float4*)(xp + 16 + 8 * half);
        float4 f3 = *(const float4*)(xp + 20 + 8 * half);
        v16bf a;
        a[0]  = f2bf(f0.x); a[1]  = f2bf(f0.y); a[2]  = f2bf(f0.z); a[3]  = f2bf(f0.w);
        a[4]  = f2bf(f1.x); a[5]  = f2bf(f1.y); a[6]  = f2bf(f1.z); a[7]  = f2bf(f1.w);
        a[8]  = f2bf(f2.x); a[9]  = f2bf(f2.y); a[10] = f2bf(f2.z); a[11] = f2bf(f2.w);
        a[12] = f2bf(f3.x); a[13] = f2bf(f3.y); a[14] = f2bf(f3.z); a[15] = f2bf(f3.w);
        const __bf16* wp = Wsw + (size_t)kt * 8 * 512 + lane * 16;
#pragma unroll
        for (int nt = 0; nt < 8; ++nt) {
            v16bf b = *(const v16bf*)(wp + nt * 512);
            acc[nt] = __builtin_amdgcn_wmma_f32_16x16x32_bf16(
                false, a, false, b, (short)0, acc[nt], false, false);
        }
    }
    // D layout: lane 0-15 -> col = nt*16+lane, row = r; lane 16-31 -> row = r+8
    float* yp = Y + (size_t)row0 * 128;
#pragma unroll
    for (int nt = 0; nt < 8; ++nt) {
        int col = nt * 16 + m;
#pragma unroll
        for (int r = 0; r < 8; ++r)
            yp[(size_t)(r + 8 * half) * 128 + col] = acc[nt][r];
    }
}

// ---------------------------------------------------------------------------
// Edge kernel: radial MLP via WMMA (9->64 silu, 64->256), then gather/scatter.
// 64 threads = 2 waves/block, each wave owns 16 edges + a 16KB LDS strip.
// ---------------------------------------------------------------------------
__global__ void edge_kernel(const float* __restrict__ vectors, const float* __restrict__ lengths,
                            const float* __restrict__ edge_feats, const int* __restrict__ edge_index,
                            const float* __restrict__ h, const float* __restrict__ radial_b1,
                            const __bf16* __restrict__ w1sw, const __bf16* __restrict__ w2sw,
                            float* __restrict__ s0, float* __restrict__ s1) {
    __shared__ float lds[2][4096];
    const int lane = threadIdx.x & 31;
    const int w = threadIdx.x >> 5;
    const int m = lane & 15;
    const int half = lane >> 4;
    const int e0 = (blockIdx.x * 2 + w) * 16;
    float* L = lds[w];

    // ---- GEMM 1: rad_in(16x9 pad 32) @ W1(32x64) ----
    const int edge = e0 + m;
    v16bf a;
#pragma unroll
    for (int e = 0; e < 16; ++e) {
        int k = (e & 7) + 16 * (e >> 3) + 8 * half;
        float v = 0.0f;
        if (k < 8)       v = edge_feats[(size_t)edge * 8 + k];
        else if (k == 8) v = lengths[edge];
        a[e] = f2bf(v);
    }
    v8f z = {};
    v8f hid[4];
#pragma unroll
    for (int nt = 0; nt < 4; ++nt) {
        v16bf b = *(const v16bf*)(w1sw + (nt * 32 + lane) * 16);
        hid[nt] = __builtin_amdgcn_wmma_f32_16x16x32_bf16(
            false, a, false, b, (short)0, z, false, false);
    }
    // bias + silu, stage hidden 16x64 in LDS
#pragma unroll
    for (int nt = 0; nt < 4; ++nt) {
        int col = nt * 16 + m;
        float bias = radial_b1[col];
#pragma unroll
        for (int r = 0; r < 8; ++r)
            L[(r + 8 * half) * 64 + col] = silu(hid[nt][r] + bias);
    }
    __syncthreads();

    // ---- reload hidden in A layout ----
    v16bf a2[2];
#pragma unroll
    for (int kt = 0; kt < 2; ++kt)
#pragma unroll
        for (int e = 0; e < 16; ++e) {
            int k = kt * 32 + (e & 7) + 16 * (e >> 3) + 8 * half;
            a2[kt][e] = f2bf(L[m * 64 + k]);
        }
    __syncthreads();

    // ---- GEMM 2: hidden(16x64) @ W2(64x256) ----
    v8f wacc[16] = {};
#pragma unroll
    for (int kt = 0; kt < 2; ++kt)
#pragma unroll
        for (int nt = 0; nt < 16; ++nt) {
            v16bf b = *(const v16bf*)(w2sw + ((kt * 16 + nt) * 32 + lane) * 16);
            wacc[nt] = __builtin_amdgcn_wmma_f32_16x16x32_bf16(
                false, a2[kt], false, b, (short)0, wacc[nt], false, false);
        }
    // stage w0||w1 (16 edges x 256) in LDS
#pragma unroll
    for (int nt = 0; nt < 16; ++nt) {
        int col = nt * 16 + m;
#pragma unroll
        for (int r = 0; r < 8; ++r)
            L[(r + 8 * half) * 256 + col] = wacc[nt][r];
    }
    __syncthreads();

    // ---- scatter: m0 = w0*hs, m1 = w1*hs*Y1; atomic add into s0/s1 (/16) ----
    const float inv_avg = 1.0f / 16.0f;
    for (int t = 0; t < 16; ++t) {
        int e = e0 + t;
        int snd = edge_index[e];
        int rcv = edge_index[NEDGES + e];
        float vx = vectors[e * 3 + 0];
        float vy = vectors[e * 3 + 1];
        float vz = vectors[e * 3 + 2];
        float nrm = sqrtf(vx * vx + vy * vy + vz * vz) + 1e-9f;
        float sc = 1.7320508075688772f / nrm;   // sqrt(3)/|v|
        float Y0 = vx * sc, Y1v = vy * sc, Y2 = vz * sc;
#pragma unroll
        for (int cc = 0; cc < 4; ++cc) {
            int c = lane + cc * 32;
            float hsv = h[(size_t)snd * 128 + c];
            float w0v = L[t * 256 + c];
            float w1v = L[t * 256 + 128 + c];
            float m0 = w0v * hsv * inv_avg;
            float mm = w1v * hsv * inv_avg;
            atomicAdd(&s0[(size_t)rcv * 128 + c], m0);
            atomicAdd(&s1[((size_t)rcv * 3 + 0) * 128 + c], mm * Y0);
            atomicAdd(&s1[((size_t)rcv * 3 + 1) * 128 + c], mm * Y1v);
            atomicAdd(&s1[((size_t)rcv * 3 + 2) * 128 + c], mm * Y2);
        }
    }
}

// ---------------------------------------------------------------------------
// Species-weighted tensor-product paths: b0, b1 from a0, a1.
// ---------------------------------------------------------------------------
__global__ void paths_kernel(const float* __restrict__ a0, const float* __restrict__ a1,
                             const float* __restrict__ node_attrs,
                             const float* __restrict__ Wprod0, const float* __restrict__ Wprod1,
                             float* __restrict__ b0, float* __restrict__ b1, int total) {
    int tid = blockIdx.x * blockDim.x + threadIdx.x;
    if (tid >= total) return;
    int n = tid >> 7, c = tid & 127;
    float a0v = a0[tid];
    float a10 = a1[((size_t)n * 3 + 0) * 128 + c];
    float a11 = a1[((size_t)n * 3 + 1) * 128 + c];
    float a12 = a1[((size_t)n * 3 + 2) * 128 + c];
    float dot = a10 * a10 + a11 * a11 + a12 * a12;
    float p0[5] = {a0v, a0v * a0v, dot, a0v * a0v * a0v, a0v * dot};
    float q[4]  = {1.0f, a0v, a0v * a0v, dot};
    float b0a = 0.0f, b1a0 = 0.0f, b1a1 = 0.0f, b1a2 = 0.0f;
    for (int s = 0; s < 10; ++s) {
        float wsp = node_attrs[n * 10 + s];
#pragma unroll
        for (int p = 0; p < 5; ++p)
            b0a += wsp * Wprod0[(s * 5 + p) * 128 + c] * p0[p];
#pragma unroll
        for (int p = 0; p < 4; ++p) {
            float t = wsp * Wprod1[(s * 4 + p) * 128 + c] * q[p];
            b1a0 += t * a10; b1a1 += t * a11; b1a2 += t * a12;
        }
    }
    b0[tid] = b0a;
    b1[((size_t)n * 3 + 0) * 128 + c] = b1a0;
    b1[((size_t)n * 3 + 1) * 128 + c] = b1a1;
    b1[((size_t)n * 3 + 2) * 128 + c] = b1a2;
}

// ---------------------------------------------------------------------------
// Readout: mh = silu(h0@Wr1); scalars = mh@Wr2; gate = silu(mh@Wg); vec = (h1@wv)*gate
// ---------------------------------------------------------------------------
__global__ void readout_kernel(const float* __restrict__ h0, const float* __restrict__ h1,
                               const float* __restrict__ Wr1, const float* __restrict__ Wr2,
                               const float* __restrict__ Wg, const float* __restrict__ wv,
                               float* __restrict__ out_scalars, float* __restrict__ out_vec) {
    int n = blockIdx.x * blockDim.x + threadIdx.x;
    if (n >= NNODES) return;
    float mh[16];
#pragma unroll
    for (int mm = 0; mm < 16; ++mm) mh[mm] = 0.0f;
    for (int k = 0; k < 128; ++k) {
        float hv = h0[(size_t)n * 128 + k];
#pragma unroll
        for (int mm = 0; mm < 16; ++mm) mh[mm] += hv * Wr1[k * 16 + mm];
    }
    float gate = 0.0f;
#pragma unroll
    for (int mm = 0; mm < 16; ++mm) {
        mh[mm] = silu(mh[mm]);
        gate += mh[mm] * Wg[mm];
    }
    gate = silu(gate);
    for (int c = 0; c < 128; ++c) {
        float sacc = 0.0f;
#pragma unroll
        for (int mm = 0; mm < 16; ++mm) sacc += mh[mm] * Wr2[mm * 128 + c];
        out_scalars[(size_t)n * 128 + c] = sacc;
    }
#pragma unroll
    for (int i = 0; i < 3; ++i) {
        float v = 0.0f;
        for (int k = 0; k < 128; ++k) v += h1[((size_t)n * 3 + i) * 128 + k] * wv[k];
        out_vec[n * 3 + i] = v * gate;
    }
}

__global__ void pack_nf(const float* __restrict__ h0, const float* __restrict__ h1,
                        float* __restrict__ nf, int total) {
    int tid = blockIdx.x * blockDim.x + threadIdx.x;
    if (tid >= total) return;
    int n = tid / 640, j = tid % 640;
    nf[tid] = (j < 128) ? h0[(size_t)n * 128 + j] : h1[(size_t)n * 384 + (j - 128)];
}

// ---------------------------------------------------------------------------
extern "C" void kernel_launch(void* const* d_in, const int* in_sizes, int n_in,
                              void* d_out, int out_size, void* d_ws, size_t ws_size,
                              hipStream_t stream) {
    const float* vectors    = (const float*)d_in[0];
    const float* lengths    = (const float*)d_in[1];
    const float* node_feats = (const float*)d_in[2];
    const float* node_attrs = (const float*)d_in[3];
    const float* edge_feats = (const float*)d_in[4];
    const int*   edge_index = (const int*)d_in[5];
    const float* W_up       = (const float*)d_in[6];
    const float* radial_w1  = (const float*)d_in[7];
    const float* radial_b1  = (const float*)d_in[8];
    const float* radial_w2  = (const float*)d_in[9];
    const float* Wd0        = (const float*)d_in[10];
    const float* Wd1        = (const float*)d_in[11];
    const float* Wprod0     = (const float*)d_in[12];
    const float* Wprod1     = (const float*)d_in[13];
    const float* Wp0        = (const float*)d_in[14];
    const float* Wp1        = (const float*)d_in[15];
    const float* Wr1        = (const float*)d_in[16];
    const float* Wr2        = (const float*)d_in[17];
    const float* Wg         = (const float*)d_in[18];
    const float* wv         = (const float*)d_in[19];

    float* ws = (float*)d_ws;
    float* R1 = ws;                       // N*128   : h -> a0 -> h0
    float* R2 = ws + 2560000;             // N*128   : s0 -> b0
    float* R3 = ws + 5120000;             // N*3*128 : s1 -> b1
    float* R4 = ws + 12800000;            // N*3*128 : a1 -> h1
    __bf16* wb = (__bf16*)(ws + 20480000);

    float* out_scalars = (float*)d_out;                 // N x 128
    float* out_vec     = out_scalars + NNODES * 128;    // N x 3
    float* out_nf      = out_vec + NNODES * 3;          // N x 640

    // 1. weight pre-swizzle (bf16 B-operand layouts)
    prep_weights<<<32, 256, 0, stream>>>(W_up, Wd0, Wd1, Wp0, Wp1, radial_w1, radial_w2, wb);
    // 2. zero scatter accumulators s0 (R2) + s1 (R3), contiguous
    zero_kernel<<<2048, 256, 0, stream>>>(R2, 2560000 + 7680000);
    // 3. h = node_feats @ W_up
    gemm128<<<157, 256, 0, stream>>>(node_feats, wb + 0, R1, NNODES);
    // 4. radial MLP + gather/scatter into s0/s1
    edge_kernel<<<NEDGES / 32, 64, 0, stream>>>(vectors, lengths, edge_feats, edge_index,
                                                R1, radial_b1, wb + 81920, wb + 83968, R2, R3);
    // 5. a0 = s0 @ Wd0 ; a1 = s1 @ Wd1
    gemm128<<<157, 256, 0, stream>>>(R2, wb + 16384, R1, NNODES);
    gemm128<<<469, 256, 0, stream>>>(R3, wb + 32768, R4, 3 * NNODES);
    // 6. species-weighted paths -> b0 (R2), b1 (R3)
    paths_kernel<<<(NNODES * 128 + 255) / 256, 256, 0, stream>>>(R1, R4, node_attrs,
                                                                 Wprod0, Wprod1, R2, R3,
                                                                 NNODES * 128);
    // 7. h0 = b0 @ Wp0 ; h1 = b1 @ Wp1
    gemm128<<<157, 256, 0, stream>>>(R2, wb + 49152, R1, NNODES);
    gemm128<<<469, 256, 0, stream>>>(R3, wb + 65536, R4, 3 * NNODES);
    // 8. readout + output packing
    readout_kernel<<<(NNODES + 255) / 256, 256, 0, stream>>>(R1, R4, Wr1, Wr2, Wg, wv,
                                                             out_scalars, out_vec);
    pack_nf<<<(NNODES * 640 + 255) / 256, 256, 0, stream>>>(R1, R4, out_nf, NNODES * 640);
}
